// RNNLayer_71253507441408
// MI455X (gfx1250) — matrix operand order repaired and checked by reading, not compiled, not run on hardware
//
#include <hip/hip_runtime.h>
#include <hip/hip_bf16.h>
#include <math.h>

// Problem constants (match reference)
#define BDIM 64
#define NDIM 2048
#define SDIM 1024
#define DDIM 128      // input feature dim (layer0) == 2H (layer1)
#define HDIM 64       // hidden per direction
#define G3H  192      // 3*H
#define NLAYERS 2

typedef __attribute__((ext_vector_type(16))) _Float16 v16h;
typedef __attribute__((ext_vector_type(8)))  float    v8f;

// ---------------------------------------------------------------------------
// A fragment from global memory: 16x32 f16 (MxK). Each lane supplies its own
// row pointer; two contiguous 16B chunks -> 2x global_load_b128.
// Layout (ISA 7.12.2): lane&15 = M; lane>>4 picks K halves {0..7,16..23} vs
// {8..15,24..31}.
// ---------------------------------------------------------------------------
__device__ __forceinline__ v16h load_a_rowptr(const _Float16* rowp, int lane) {
    const _Float16* p = rowp + ((lane >> 4) << 3);
    v16h a;
#pragma unroll
    for (int i = 0; i < 8; ++i) a[i] = p[i];
#pragma unroll
    for (int i = 0; i < 8; ++i) a[8 + i] = p[16 + i];
    return a;
}

__device__ __forceinline__ float sigmoidf_(float x) {
    return 1.0f / (1.0f + __expf(-x));
}

// ---------------------------------------------------------------------------
// Kernel 1: scatter node states into leaf sequence slots + count lengths.
// grid (N, B), block 128.
// ---------------------------------------------------------------------------
__global__ void gather_kernel(const float* __restrict__ states,
                              const int* __restrict__ pos,
                              _Float16* __restrict__ seqA,
                              int* __restrict__ lengths) {
    int n = blockIdx.x, b = blockIdx.y;
    int p = pos[b * NDIM + n];
    if (p > 0) {
        if (threadIdx.x == 0) atomicAdd(&lengths[b], 1);
        const float* src = states + ((size_t)b * NDIM + n) * DDIM;
        _Float16* dst = seqA + ((size_t)b * SDIM + (p - 1)) * DDIM;
        dst[threadIdx.x] = (_Float16)src[threadIdx.x];
    }
}

// ---------------------------------------------------------------------------
// Kernel 2: xp[dir] = X(dir-order) @ W[dir] + bi[dir]
// grid (512, 2): x = 128-row tile over B*S, y = dir. block 256 = 8 waves.
// W staged in LDS pre-swizzled in WMMA B-fragment layout.
// amdgpu_waves_per_eu(1): allow up to 256 VGPRs so the 12 B fragments of a
// k-chunk can be clause-loaded in flight instead of dscnt-0 serializing.
// ---------------------------------------------------------------------------
__global__ __launch_bounds__(256)
__attribute__((amdgpu_waves_per_eu(1)))
void xw_gemm_kernel(const _Float16* __restrict__ seqA,
                    const float* __restrict__ Wf_l, const float* __restrict__ Wb_l,
                    const float* __restrict__ bif,  const float* __restrict__ bib,
                    const int* __restrict__ lengths,
                    float* __restrict__ xp) {
    __shared__ v16h  Wsw[4][12][32];   // 48 KB, fragment layout
    __shared__ float bl[G3H];

    const int dir = blockIdx.y;
    const float* W  = dir ? Wb_l : Wf_l;
    const float* bi = dir ? bib  : bif;

    // Fill pre-swizzled weights: 4*12*32 = 1536 fragment-lanes, 6 per thread.
    for (int fl = threadIdx.x; fl < 4 * 12 * 32; fl += 256) {
        int ln = fl & 31;
        int nt = (fl >> 5) % 12;
        int kc = fl / (12 * 32);
        int n  = nt * 16 + (ln & 15);
        int k0 = kc * 32 + ((ln >> 4) << 3);
        v16h v;
#pragma unroll
        for (int h = 0; h < 16; ++h) v[h] = (_Float16)W[(size_t)(k0 + (h & 7) + ((h >> 3) << 4)) * G3H + n];
        Wsw[kc][nt][ln] = v;
    }
    for (int i = threadIdx.x; i < G3H; i += 256) bl[i] = bi[i];
    __syncthreads();

    const int wave = threadIdx.x >> 5;
    const int lane = threadIdx.x & 31;

    // Per-lane source row (handles bwd ragged time reversal).
    const int m = blockIdx.x * 128 + wave * 16 + (lane & 15);
    const int b = m >> 10;             // /S
    const int t = m & (SDIM - 1);      // %S
    int tsrc = t;
    if (dir) {
        int len = lengths[b];
        tsrc = (t < len) ? (len - 1 - t) : t;
    }
    const _Float16* arow = seqA + ((size_t)b * SDIM + tsrc) * DDIM;

    v8f acc[12];
#pragma unroll
    for (int nt = 0; nt < 12; ++nt) acc[nt] = (v8f){};

    // Software-pipelined A (global) and double-buffered B groups (LDS).
    v16h a_next = load_a_rowptr(arow, lane);
#pragma unroll
    for (int kc = 0; kc < 4; ++kc) {
        v16h a = a_next;
        if (kc < 3) a_next = load_a_rowptr(arow + (kc + 1) * 32, lane);

        v16h bw0[6], bw1[6];
#pragma unroll
        for (int q = 0; q < 6; ++q) bw0[q] = Wsw[kc][q][lane];
#pragma unroll
        for (int q = 0; q < 6; ++q) bw1[q] = Wsw[kc][6 + q][lane];
#pragma unroll
        for (int q = 0; q < 6; ++q) {
            acc[q] = __builtin_amdgcn_wmma_f32_16x16x32_f16(
                false, a, false, bw0[q], (short)0, acc[q], false, false);
        }
#pragma unroll
        for (int q = 0; q < 6; ++q) {
            acc[6 + q] = __builtin_amdgcn_wmma_f32_16x16x32_f16(
                false, a, false, bw1[q], (short)0, acc[6 + q], false, false);
        }
    }

    // Epilogue: add input bias, store f32. D layout: row = i + 8*(lane>>4).
    float* xpd = xp + (size_t)dir * ((size_t)BDIM * SDIM * G3H);
    const int mrow0 = blockIdx.x * 128 + wave * 16 + 8 * (lane >> 4);
    const int colL  = lane & 15;
#pragma unroll
    for (int nt = 0; nt < 12; ++nt) {
        int col = nt * 16 + colL;
        float bias = bl[col];
#pragma unroll
        for (int i = 0; i < 8; ++i) {
            xpd[(size_t)(mrow0 + i) * G3H + col] = acc[nt][i] + bias;
        }
    }
}

// ---------------------------------------------------------------------------
// Kernel 3: recurrent scan. grid 8 blocks: blockIdx.x = dir*4 + batch-quarter.
// block 128 = 4 waves; wave w owns N tiles 3w..3w+2 of hp = h @ U (16x192).
// U fragments are loop-invariant -> register-resident for all 1024 steps.
// Per-step LDS traffic: 2 A-frag loads + 3 v8f accumulator stores + gates.
// ---------------------------------------------------------------------------
__global__ __launch_bounds__(128)
__attribute__((amdgpu_waves_per_eu(1)))
void scan_kernel(const float* __restrict__ xp,
                 const float* __restrict__ Uf_l, const float* __restrict__ Ub_l,
                 const float* __restrict__ brf,  const float* __restrict__ brb,
                 const int* __restrict__ lengths,
                 float* __restrict__ seq_next,
                 _Float16* __restrict__ seqA_next) {
    __shared__ v16h  Usw[2][12][32];   // 24 KB (staging for register hoist)
    __shared__ v16h  h16sw[2][32];     // 2 KB  (WMMA A operand, frag layout)
    __shared__ float h32[16][HDIM];    // 4 KB  (f32 master state)
    __shared__ v8f   hpsw[12][32];     // 12 KB (h @ U accum, frag layout)
    __shared__ float brl[G3H];

    const int dir = blockIdx.x >> 2;
    const int bq  = blockIdx.x & 3;
    const float* U  = dir ? Ub_l : Uf_l;
    const float* br = dir ? brb  : brf;

    // Fill pre-swizzled U: 2*12*32 = 768 fragment-lanes, 6 per thread.
    for (int fl = threadIdx.x; fl < 2 * 12 * 32; fl += 128) {
        int ln = fl & 31;
        int nt = (fl >> 5) % 12;
        int kc = fl / (12 * 32);
        int n  = nt * 16 + (ln & 15);
        int k0 = kc * 32 + ((ln >> 4) << 3);
        v16h v;
#pragma unroll
        for (int h = 0; h < 16; ++h) v[h] = (_Float16)U[(size_t)(k0 + (h & 7) + ((h >> 3) << 4)) * G3H + n];
        Usw[kc][nt][ln] = v;
    }
    for (int i = threadIdx.x; i < G3H; i += 128) brl[i] = br[i];
    if (threadIdx.x < 64) h16sw[threadIdx.x >> 5][threadIdx.x & 31] = (v16h){};
    for (int i = threadIdx.x; i < 16 * HDIM; i += 128) (&h32[0][0])[i] = 0.0f;
    __syncthreads();

    const int wave = threadIdx.x >> 5;
    const int lane = threadIdx.x & 31;
    const float* xpd = xp + (size_t)dir * ((size_t)BDIM * SDIM * G3H);

    // Hoist this wave's 6 U fragments into registers (48 VGPRs, persistent).
    v16h bu[3][2];
#pragma unroll
    for (int nt3 = 0; nt3 < 3; ++nt3)
#pragma unroll
        for (int kc = 0; kc < 2; ++kc)
            bu[nt3][kc] = Usw[kc][wave * 3 + nt3][lane];

    for (int t = 0; t < SDIM; ++t) {
        // --- hp = h @ U (WMMA); A fragments loaded once per step ---
        v16h a0 = h16sw[0][lane];
        v16h a1 = h16sw[1][lane];
#pragma unroll
        for (int nt3 = 0; nt3 < 3; ++nt3) {
            v8f acc = (v8f){};
            acc = __builtin_amdgcn_wmma_f32_16x16x32_f16(
                false, a0, false, bu[nt3][0], (short)0, acc, false, false);
            acc = __builtin_amdgcn_wmma_f32_16x16x32_f16(
                false, a1, false, bu[nt3][1], (short)0, acc, false, false);
            hpsw[wave * 3 + nt3][lane] = acc;      // 2x ds_store_b128
        }
        __syncthreads();

        // --- gate epilogue: 16 rows x 64 units, 8 per thread ---
#pragma unroll
        for (int q = 0; q < 8; ++q) {
            const int idx = threadIdx.x + q * 128;   // 0..1023
            const int r   = idx >> 6;
            const int j   = idx & 63;
            const int bg  = bq * 16 + r;
            const size_t mrow = (size_t)bg * SDIM + t;
            const float* xrow = xpd + mrow * G3H;

            // hp reads via inverse C/D fragment mapping:
            //   lane = (col&15) | ((row>>3)<<4), element = row&7
            const int li = ((r >> 3) << 4);
            const int ei = r & 7;
            float hz = ((const float*)&hpsw[(j) >> 4][((j) & 15) | li])[ei]
                       + brl[j];
            float hr = ((const float*)&hpsw[(64 + j) >> 4][((64 + j) & 15) | li])[ei]
                       + brl[64 + j];
            float hh = ((const float*)&hpsw[(128 + j) >> 4][((128 + j) & 15) | li])[ei]
                       + brl[128 + j];

            float z    = sigmoidf_(xrow[j] + hz);
            float rr   = sigmoidf_(xrow[64 + j] + hr);
            float cand = tanhf(xrow[128 + j] + rr * hh);
            float hprev = h32[r][j];
            float hnew  = z * hprev + (1.0f - z) * cand;
            if (t >= lengths[bg]) hnew = hprev;   // freeze past row length

            h32[r][j] = hnew;
            // write h into A-fragment layout: K=j, M=r
            //   lanehi = (j>>3)&1 ; half = (j&7) + ((j>>4)<<3)
            {
                const int kc     = j >> 5;
                const int k32    = j & 31;
                const int lanehi = (k32 >> 3) & 1;
                const int half   = (k32 & 7) + ((k32 >> 4) << 3);
                ((_Float16*)&h16sw[kc][(lanehi << 4) | r])[half] = (_Float16)hnew;
            }
            const size_t o = mrow * DDIM + (size_t)dir * HDIM + j;
            seq_next[o]  = hnew;
            seqA_next[o] = (_Float16)hnew;   // next layer's GEMM input
        }
        __syncthreads();
    }
}

// ---------------------------------------------------------------------------
// Kernel 4: scatter leaf sequence back into node states. grid (N, B), block 128.
// ---------------------------------------------------------------------------
__global__ void scatter_kernel(const float* __restrict__ states,
                               const int* __restrict__ pos,
                               const float* __restrict__ seq,
                               float* __restrict__ out) {
    int n = blockIdx.x, b = blockIdx.y, d = threadIdx.x;
    int p = pos[b * NDIM + n];
    size_t o = ((size_t)b * NDIM + n) * DDIM + d;
    out[o] = (p > 0) ? seq[((size_t)b * SDIM + (p - 1)) * DDIM + d] : states[o];
}

// ---------------------------------------------------------------------------
extern "C" void kernel_launch(void* const* d_in, const int* in_sizes, int n_in,
                              void* d_out, int out_size, void* d_ws, size_t ws_size,
                              hipStream_t stream) {
    const float* states = (const float*)d_in[0];
    const int*   pos    = (const int*)  d_in[1];
    const float* Wf     = (const float*)d_in[2];
    const float* Uf     = (const float*)d_in[3];
    const float* bf     = (const float*)d_in[4];
    const float* Wb     = (const float*)d_in[5];
    const float* Ub     = (const float*)d_in[6];
    const float* bb     = (const float*)d_in[7];
    float* out = (float*)d_out;

    // Workspace layout
    char* ws = (char*)d_ws;
    int*      lengths = (int*)ws;                                  // 1 KB
    size_t off = 1024;
    _Float16* seqA = (_Float16*)(ws + off);                        // 16 MB
    off += (size_t)BDIM * SDIM * DDIM * sizeof(_Float16);
    float*    xp   = (float*)(ws + off);                           // 100 MB
    off += (size_t)2 * BDIM * SDIM * G3H * sizeof(float);
    float*    seqN = (float*)(ws + off);                           // 32 MB

    hipMemsetAsync(lengths, 0, 1024, stream);
    hipMemsetAsync(seqA, 0, (size_t)BDIM * SDIM * DDIM * sizeof(_Float16), stream);

    gather_kernel<<<dim3(NDIM, BDIM), DDIM, 0, stream>>>(states, pos, seqA, lengths);

    for (int l = 0; l < NLAYERS; ++l) {
        const float* Wf_l = Wf + (size_t)l * DDIM * G3H;
        const float* Wb_l = Wb + (size_t)l * DDIM * G3H;
        const float* Uf_l = Uf + (size_t)l * HDIM * G3H;
        const float* Ub_l = Ub + (size_t)l * HDIM * G3H;
        const float* bif  = bf + (size_t)l * 2 * G3H;      // b[l][0]
        const float* brf  = bif + G3H;                     // b[l][1]
        const float* bib  = bb + (size_t)l * 2 * G3H;
        const float* brb  = bib + G3H;

        xw_gemm_kernel<<<dim3(512, 2), 256, 0, stream>>>(
            seqA, Wf_l, Wb_l, bif, bib, lengths, xp);

        scan_kernel<<<8, 128, 0, stream>>>(
            xp, Uf_l, Ub_l, brf, brb, lengths, seqN, seqA);
    }

    scatter_kernel<<<dim3(NDIM, BDIM), DDIM, 0, stream>>>(states, pos, seqN, out);
}
